// MainNet_35519379538315
// MI455X (gfx1250) — matrix operand order repaired and checked by reading, not compiled, not run on hardware
//
#include <hip/hip_runtime.h>
#include <math.h>

typedef __attribute__((ext_vector_type(16))) _Float16 v16h;
typedef __attribute__((ext_vector_type(8)))  float    v8f;

static constexpr int B_ = 16;
static constexpr int N_ = 2048;
static constexpr int T_ = 8192;
static constexpr int K_ = 17;
static constexpr int ROWS   = B_*N_*K_;   // 557056 (divisible by 16)
static constexpr int MTILES = ROWS/16;    // 34816
static constexpr int P_     = B_*N_;      // 32768
static constexpr int PTILES = P_/16;      // 2048

// ---------------- small scalar kernels ----------------

__global__ void gather3_k(const float* __restrict__ in, const int* __restrict__ didx,
                          int level, float* __restrict__ out) {
  int t = blockIdx.x*blockDim.x + threadIdx.x;
  if (t >= P_) return;
  int b = t >> 11, n = t & (N_-1);
  int j = didx[b*T_ + level*N_ + n];
  const float* s = in + (size_t)(b*N_ + j)*3;
  float* d = out + (size_t)t*3;
  d[0]=s[0]; d[1]=s[1]; d[2]=s[2];
}

__global__ void lc_compute_k(const float* __restrict__ cur, const float* __restrict__ axes,
                             const int* __restrict__ neigh, int level, float* __restrict__ lcb) {
  int t = blockIdx.x*blockDim.x + threadIdx.x;
  if (t >= ROWS) return;
  int k = t % K_; int bn = t / K_;
  int b = bn >> 11, n = bn & (N_-1);
  int nb = neigh[(size_t)(b*T_ + level*N_ + n)*33 + k];
  const float* pc = cur + (size_t)(b*N_ + n)*3;
  const float* pq = cur + (size_t)(b*N_ + nb)*3;
  float g0 = pq[0]-pc[0], g1 = pq[1]-pc[1], g2 = pq[2]-pc[2];
  const float* A = axes + (size_t)(b*T_ + level*N_ + n)*9;
  float* d = lcb + (size_t)t*3;
  d[0] = g0*A[0] + g1*A[3] + g2*A[6];
  d[1] = g0*A[1] + g1*A[4] + g2*A[7];
  d[2] = g0*A[2] + g1*A[5] + g2*A[8];
}

__global__ void lc_cons_k(const float* __restrict__ axes, const int* __restrict__ neigh,
                          int level, float* __restrict__ acc) {
  int t = blockIdx.x*blockDim.x + threadIdx.x;
  if (t >= P_) return;
  int b = t >> 11, n = t & (N_-1);
  float xs[K_][3], ys[K_][3];
  for (int k = 0; k < K_; ++k) {
    int nb = neigh[(size_t)(b*T_ + level*N_ + n)*33 + k];
    const float* A = axes + (size_t)(b*T_ + level*N_ + nb)*9;
    xs[k][0]=A[0]; xs[k][1]=A[3]; xs[k][2]=A[6];
    ys[k][0]=A[1]; ys[k][1]=A[4]; ys[k][2]=A[7];
  }
  float thr = cosf(15.f*(float)(level+1)*3.14159265358979f/180.f);
  float sx=0.f, cx=0.f, sy=0.f, cy=0.f;
  for (int i = 0; i < K_; ++i)
    for (int j = 0; j < K_; ++j) {
      float dx = xs[i][0]*xs[j][0]+xs[i][1]*xs[j][1]+xs[i][2]*xs[j][2];
      if (dx < thr) { sx += dx; cx += 1.f; }
      float dy = ys[i][0]*ys[j][0]+ys[i][1]*ys[j][1]+ys[i][2]*ys[j][2];
      if (dy < thr) { sy += dy; cy += 1.f; }
    }
  atomicAdd(&acc[level*4+0], sx);
  atomicAdd(&acc[level*4+1], cx);
  atomicAdd(&acc[level*4+2], sy);
  atomicAdd(&acc[level*4+3], cy);
}

// Pack conv weight (CIN,COUT) f32 -> half [co][CPAD] (transposed, zero-padded K)
__global__ void conv_w_prep_k(const float* __restrict__ W, _Float16* __restrict__ Wh,
                              int CIN, int CPAD, int COUT) {
  int co = blockIdx.x*blockDim.x + threadIdx.x;
  if (co >= COUT) return;
  for (int k = 0; k < CPAD; ++k)
    Wh[(size_t)co*CPAD + k] = (_Float16)((k < CIN) ? W[(size_t)k*COUT + co] : 0.f);
}

// Weight-norm FC weight (COUT,CIN) f32 -> half [co][CPAD], rows >= COUT zeroed
__global__ void fc_w_prep_k(const float* __restrict__ v, const float* __restrict__ g,
                            _Float16* __restrict__ Wh, int CIN, int CPAD, int COUT, int COPAD) {
  int co = blockIdx.x*blockDim.x + threadIdx.x;
  if (co >= COPAD) return;
  if (co >= COUT) {
    for (int k = 0; k < CPAD; ++k) Wh[(size_t)co*CPAD + k] = (_Float16)0.f;
    return;
  }
  float s = 0.f;
  for (int i = 0; i < CIN; ++i) { float w = v[(size_t)co*CIN + i]; s += w*w; }
  float sc = g[co] / sqrtf(s);
  for (int k = 0; k < CPAD; ++k)
    Wh[(size_t)co*CPAD + k] = (_Float16)((k < CIN) ? v[(size_t)co*CIN + k]*sc : 0.f);
}

// ---------------- WMMA surface-conv GEMM ----------------
// One wave = one 16-row tile x NBT adjacent 16-col tiles of the flattened
// (B*N*17) x COUT GEMM. The gathered A fragment is built once per K-slice and
// reused for NBT back-to-back WMMAs. Channel order: [points(PDIM), lc(3), rel(3), pad].
template<int PDIM, int CIN, int CPAD, int COUT, int NBT>
__global__ __launch_bounds__(256)
void conv_wmma_k(const float* __restrict__ points, const float* __restrict__ lcb,
                 const float* __restrict__ scx, const int* __restrict__ neigh,
                 const int* __restrict__ didx, int level,
                 const _Float16* __restrict__ Wh, const float* __restrict__ bias,
                 unsigned* __restrict__ outb)
{
  constexpr int NGRP = COUT/16/NBT;   // column-tile groups
  int lane = threadIdx.x & 31;
  int wave = threadIdx.x >> 5;
  int wid  = blockIdx.x*8 + wave;
  int tile_m = wid / NGRP;
  int cg     = wid - tile_m*NGRP;
  int l15 = lane & 15;
  int gr = tile_m*16 + l15;
  int b   = gr / (N_*K_);
  int rem = gr - b*(N_*K_);
  int n = rem / K_;
  int k = rem - n*K_;
  int nb   = neigh[(size_t)(b*T_ + level*N_ + n)*33 + k];
  int pidx = didx[b*T_ + level*N_ + nb];
  const float* lcp = lcb + (size_t)((b*N_ + n)*K_ + k)*3;
  float l0 = lcp[0], l1 = lcp[1], l2 = lcp[2];
  const float* pc = scx + (size_t)(b*N_ + n)*3;
  const float* pq = scx + (size_t)(b*N_ + nb)*3;
  float rx = pq[0]-pc[0], ry = pq[1]-pc[1], rz = pq[2]-pc[2];

  int gA = (lane & 16) ? 16 : 0;    // K sub-offset per A-layout lane group
  int kh = (lane & 16) ? 16 : 0;    // K sub-offset per B-layout lane group
  const _Float16* Wbase = Wh + (size_t)(cg*NBT*16 + l15)*CPAD + kh;

  v8f acc[NBT] = {};
  #pragma unroll
  for (int kc = 0; kc < CPAD/32; ++kc) {
    v16h a;
    #pragma unroll
    for (int i = 0; i < 16; ++i) {
      int c = kc*32 + gA + i;   // K = gA..gA+15
      float v;
      if (c < PDIM) {
        v = points[(size_t)(b*N_ + pidx)*PDIM + c];
      } else {
        int cc = c - PDIM;
        v = (cc==0)?l0:(cc==1)?l1:(cc==2)?l2:(cc==3)?rx:(cc==4)?ry:(cc==5)?rz:0.f;
      }
      a[i] = (_Float16)v;
    }
    #pragma unroll
    for (int j = 0; j < NBT; ++j) {
      v16h bf = *reinterpret_cast<const v16h*>(Wbase + (size_t)j*16*CPAD + kc*32);
      acc[j] = __builtin_amdgcn_wmma_f32_16x16x32_f16(false, a, false, bf, (short)0, acc[j], false, false);
    }
  }
  int Mb = (lane & 16) ? 8 : 0;
  #pragma unroll
  for (int j = 0; j < NBT; ++j) {
    int ncol = (cg*NBT + j)*16 + l15;
    float bv = bias[ncol];
    #pragma unroll
    for (int r = 0; r < 8; ++r) {
      int g2 = tile_m*16 + Mb + r;
      int b2 = g2 / (N_*K_);
      int r2 = g2 - b2*(N_*K_);
      int n2 = r2 / K_;
      float val = fmaxf(acc[j][r] + bv, 0.f);        // relu; max over k via atomic on fp bits
      atomicMax(outb + (size_t)(b2*N_ + n2)*COUT + ncol, __float_as_uint(val));
    }
  }
}

// ---------------- WMMA FC GEMM ----------------
// MODE 0: A = [l2p f32 (256) | cls_label (16) | pad]; MODE 1: A = half buffer, pitch CPAD.
template<int MODE, int CPAD, int COUT, int NVALID, bool RELU, int NBT>
__global__ __launch_bounds__(256)
void fc_wmma_k(const float* __restrict__ fin32, const _Float16* __restrict__ finh,
               const float* __restrict__ cls, const _Float16* __restrict__ Wh,
               const float* __restrict__ bias, _Float16* __restrict__ outh,
               float* __restrict__ out32)
{
  constexpr int NGRP = COUT/16/NBT;
  int lane = threadIdx.x & 31;
  int wave = threadIdx.x >> 5;
  int wid  = blockIdx.x*8 + wave;
  int tile_m = wid / NGRP;
  int cg     = wid - tile_m*NGRP;
  int l15 = lane & 15;
  int p = tile_m*16 + l15;
  int b = p >> 11;
  int gA = (lane & 16) ? 16 : 0;
  int kh = (lane & 16) ? 16 : 0;
  const _Float16* Wbase = Wh + (size_t)(cg*NBT*16 + l15)*CPAD + kh;

  v8f acc[NBT] = {};
  #pragma unroll 4
  for (int kc = 0; kc < CPAD/32; ++kc) {
    v16h a;
    #pragma unroll
    for (int i = 0; i < 16; ++i) {
      int c = kc*32 + gA + i;
      float v;
      if (MODE == 0) {
        v = (c < 256) ? fin32[(size_t)p*256 + c]
                      : ((c < 272) ? cls[b*16 + (c - 256)] : 0.f);
      } else {
        v = (float)finh[(size_t)p*CPAD + c];
      }
      a[i] = (_Float16)v;
    }
    #pragma unroll
    for (int j = 0; j < NBT; ++j) {
      v16h bf = *reinterpret_cast<const v16h*>(Wbase + (size_t)j*16*CPAD + kc*32);
      acc[j] = __builtin_amdgcn_wmma_f32_16x16x32_f16(false, a, false, bf, (short)0, acc[j], false, false);
    }
  }
  int Mb = (lane & 16) ? 8 : 0;
  #pragma unroll
  for (int j = 0; j < NBT; ++j) {
    int ncol = (cg*NBT + j)*16 + l15;
    if (ncol < NVALID) {
      float bv = bias[ncol];
      #pragma unroll
      for (int r = 0; r < 8; ++r) {
        int p2 = tile_m*16 + Mb + r;
        float val = acc[j][r] + bv;
        if (RELU) val = fmaxf(val, 0.f);
        if (outh) outh[(size_t)p2*COUT + ncol] = (_Float16)val;
        else      out32[(size_t)p2*NVALID + ncol] = val;
      }
    }
  }
}

__global__ void logsoftmax_k(float* __restrict__ x) {
  int p = blockIdx.x*blockDim.x + threadIdx.x;
  if (p >= P_) return;
  float* row = x + (size_t)p*40;
  float m = row[0];
  for (int i = 1; i < 40; ++i) m = fmaxf(m, row[i]);
  float s = 0.f;
  for (int i = 0; i < 40; ++i) s += expf(row[i] - m);
  float l = logf(s);
  for (int i = 0; i < 40; ++i) row[i] = row[i] - m - l;
}

__global__ void finalize_k(const float* __restrict__ acc, float* __restrict__ outs) {
  if (threadIdx.x != 0 || blockIdx.x != 0) return;
  float s = 0.f;
  for (int L = 0; L < 4; ++L) {
    float sx=acc[L*4+0], cx=acc[L*4+1], sy=acc[L*4+2], cy=acc[L*4+3];
    s += (cx > 0.f) ? sx / fmaxf(cx, 1.f) : 0.f;
    s += (cy > 0.f) ? sy / fmaxf(cy, 1.f) : 0.f;
  }
  outs[0] = 0.f;
  outs[1] = s;
}

// ---------------- host ----------------

extern "C" void kernel_launch(void* const* d_in, const int* in_sizes, int n_in,
                              void* d_out, int out_size, void* d_ws, size_t ws_size,
                              hipStream_t stream) {
  (void)in_sizes; (void)n_in; (void)out_size; (void)ws_size;
  const float* xyz   = (const float*)d_in[0];
  const int*   neigh = (const int*)d_in[1];
  const int*   didx  = (const int*)d_in[2];
  const float* axes  = (const float*)d_in[3];
  const float* cls   = (const float*)d_in[4];
  const float* w0 =(const float*)d_in[5],  *b0 =(const float*)d_in[6];
  const float* w02=(const float*)d_in[7],  *b02=(const float*)d_in[8];
  const float* w1 =(const float*)d_in[9],  *b1 =(const float*)d_in[10];
  const float* w12=(const float*)d_in[11], *b12=(const float*)d_in[12];
  const float* w2 =(const float*)d_in[13], *b2 =(const float*)d_in[14];
  const float* f1v=(const float*)d_in[15], *f1g=(const float*)d_in[16], *f1b=(const float*)d_in[17];
  const float* f2v=(const float*)d_in[18], *f2g=(const float*)d_in[19], *f2b=(const float*)d_in[20];
  const float* f3v=(const float*)d_in[21], *f3g=(const float*)d_in[22], *f3b=(const float*)d_in[23];
  float* out = (float*)d_out;

  size_t off = 0;
  auto alloc = [&](size_t bytes) -> void* {
    off = (off + 255) & ~(size_t)255;
    void* p = (char*)d_ws + off;
    off += bytes;
    return p;
  };
  const size_t BN3 = (size_t)P_*3*4;
  float* cur1 = (float*)alloc(BN3);   // xyz[di0]  (== sc_xyz for layer 0)
  float* cur2 = (float*)alloc(BN3);
  float* cur3 = (float*)alloc(BN3);
  float* cur4 = (float*)alloc(BN3);
  float* sc02 = (float*)alloc(BN3);
  float* sc1  = (float*)alloc(BN3);
  float* sc12 = (float*)alloc(BN3);
  float* sc2  = (float*)alloc(BN3);
  const size_t LCB = (size_t)ROWS*3*4;
  float* lc0 = (float*)alloc(LCB);
  float* lc1 = (float*)alloc(LCB);
  float* lc2 = (float*)alloc(LCB);
  float* lc3 = (float*)alloc(LCB);
  unsigned* o0  = (unsigned*)alloc((size_t)P_*32*4);
  unsigned* o02 = (unsigned*)alloc((size_t)P_*32*4);
  unsigned* o1  = (unsigned*)alloc((size_t)P_*128*4);
  unsigned* o12 = (unsigned*)alloc((size_t)P_*128*4);
  unsigned* o2  = (unsigned*)alloc((size_t)P_*256*4);
  _Float16* h1 = (_Float16*)alloc((size_t)P_*512*2);
  _Float16* h2 = (_Float16*)alloc((size_t)P_*256*2);
  _Float16* w0h  = (_Float16*)alloc((size_t)32*32*2);
  _Float16* w02h = (_Float16*)alloc((size_t)32*64*2);
  _Float16* w1h  = (_Float16*)alloc((size_t)128*64*2);
  _Float16* w12h = (_Float16*)alloc((size_t)128*160*2);
  _Float16* w2h  = (_Float16*)alloc((size_t)256*160*2);
  _Float16* f1h = (_Float16*)alloc((size_t)512*288*2);
  _Float16* f2h = (_Float16*)alloc((size_t)256*512*2);
  _Float16* f3h = (_Float16*)alloc((size_t)48*256*2);
  float* accb = (float*)alloc(16*4);

  // zero atomic-max targets + loss accumulators (graph-capture safe)
  hipMemsetAsync(o0,  0, (size_t)P_*32*4,  stream);
  hipMemsetAsync(o02, 0, (size_t)P_*32*4,  stream);
  hipMemsetAsync(o1,  0, (size_t)P_*128*4, stream);
  hipMemsetAsync(o12, 0, (size_t)P_*128*4, stream);
  hipMemsetAsync(o2,  0, (size_t)P_*256*4, stream);
  hipMemsetAsync(accb, 0, 16*4, stream);

  const int GB = (P_ + 255)/256;      // 128
  const int LB = (ROWS + 255)/256;    // 2176

  // hierarchical xyz chains
  gather3_k<<<GB,256,0,stream>>>(xyz,  didx, 0, cur1);
  gather3_k<<<GB,256,0,stream>>>(cur1, didx, 1, cur2);
  gather3_k<<<GB,256,0,stream>>>(cur2, didx, 2, cur3);
  gather3_k<<<GB,256,0,stream>>>(cur3, didx, 3, cur4);
  gather3_k<<<GB,256,0,stream>>>(cur1, didx, 0, sc02);
  gather3_k<<<GB,256,0,stream>>>(sc02, didx, 1, sc1);
  gather3_k<<<GB,256,0,stream>>>(sc1,  didx, 2, sc12);
  gather3_k<<<GB,256,0,stream>>>(sc12, didx, 3, sc2);

  // local coordinates + consistency loss
  lc_compute_k<<<LB,256,0,stream>>>(cur1, axes, neigh, 0, lc0);
  lc_compute_k<<<LB,256,0,stream>>>(cur2, axes, neigh, 1, lc1);
  lc_compute_k<<<LB,256,0,stream>>>(cur3, axes, neigh, 2, lc2);
  lc_compute_k<<<LB,256,0,stream>>>(cur4, axes, neigh, 3, lc3);
  lc_cons_k<<<GB,256,0,stream>>>(axes, neigh, 0, accb);
  lc_cons_k<<<GB,256,0,stream>>>(axes, neigh, 1, accb);
  lc_cons_k<<<GB,256,0,stream>>>(axes, neigh, 2, accb);
  lc_cons_k<<<GB,256,0,stream>>>(axes, neigh, 3, accb);

  // weight packing (half, [co][K_pad])
  conv_w_prep_k<<<1,256,0,stream>>>(w0,  w0h,    6,  32,  32);
  conv_w_prep_k<<<1,256,0,stream>>>(w02, w02h,  38,  64,  32);
  conv_w_prep_k<<<1,256,0,stream>>>(w1,  w1h,   38,  64, 128);
  conv_w_prep_k<<<1,256,0,stream>>>(w12, w12h, 134, 160, 128);
  conv_w_prep_k<<<1,256,0,stream>>>(w2,  w2h,  134, 160, 256);
  fc_w_prep_k<<<2,256,0,stream>>>(f1v, f1g, f1h, 272, 288, 512, 512);
  fc_w_prep_k<<<1,256,0,stream>>>(f2v, f2g, f2h, 512, 512, 256, 256);
  fc_w_prep_k<<<1,256,0,stream>>>(f3v, f3g, f3h, 256, 256,  40,  48);

  // surface-conv stack (WMMA); grid = MTILES * (COUT/16/NBT) / 8 blocks of 8 waves
  conv_wmma_k<0,6,32,32,2>      <<<MTILES*1/8, 256,0,stream>>>(nullptr,           lc0, cur1, neigh, didx, 0, w0h,  b0,  o0);
  conv_wmma_k<32,38,64,32,2>    <<<MTILES*1/8, 256,0,stream>>>((const float*)o0,  lc0, sc02, neigh, didx, 0, w02h, b02, o02);
  conv_wmma_k<32,38,64,128,4>   <<<MTILES*2/8, 256,0,stream>>>((const float*)o02, lc1, sc1,  neigh, didx, 1, w1h,  b1,  o1);
  conv_wmma_k<128,134,160,128,4><<<MTILES*2/8, 256,0,stream>>>((const float*)o1,  lc2, sc12, neigh, didx, 2, w12h, b12, o12);
  conv_wmma_k<128,134,160,256,4><<<MTILES*4/8, 256,0,stream>>>((const float*)o12, lc3, sc2,  neigh, didx, 3, w2h,  b2,  o2);

  // FC head (WMMA) + log-softmax
  fc_wmma_k<0,288,512,512,true,4> <<<PTILES*8/8,256,0,stream>>>((const float*)o2, nullptr, cls, f1h, f1b, h1, nullptr);
  fc_wmma_k<1,512,256,256,true,4> <<<PTILES*4/8,256,0,stream>>>(nullptr, h1, nullptr, f2h, f2b, h2, nullptr);
  fc_wmma_k<1,256,48,40,false,3>  <<<PTILES*1/8,256,0,stream>>>(nullptr, h2, nullptr, f3h, f3b, nullptr, out);
  logsoftmax_k<<<GB,256,0,stream>>>(out);

  finalize_k<<<1,64,0,stream>>>(accb, out + (size_t)P_*40);
}
